// Decoder_30760555773952
// MI455X (gfx1250) — matrix operand order repaired and checked by reading, not compiled, not run on hardware
//
#include <hip/hip_runtime.h>

// Problem constants (fixed by the reference)
#define TT 64
#define BB 32
#define SS 512
#define HH 512
#define EE 512
#define VV 32000

typedef __attribute__((ext_vector_type(16))) __bf16 v16bf;
typedef __attribute__((ext_vector_type(8)))  float  v8f;

struct B32x2 { uint4 a, b; };

__device__ __forceinline__ unsigned short f32_to_bf16(float f) {
  unsigned int u = __float_as_uint(f);
  u += 0x7FFFu + ((u >> 16) & 1u);   // round-to-nearest-even
  return (unsigned short)(u >> 16);
}
__device__ __forceinline__ float sigm(float x) { return 1.0f / (1.0f + __expf(-x)); }

__device__ __forceinline__ v16bf load_frag_a(const unsigned short* __restrict__ A,
                                             size_t m, int K, int k0, int half) {
  // ISA 16-bit A 16x32 layout: lanes<16 hold K=[0..7],[16..23]; lanes>=16 hold K=[8..15],[24..31]
  const uint4* p0 = reinterpret_cast<const uint4*>(A + m * K + k0 + half * 8);
  const uint4* p1 = reinterpret_cast<const uint4*>(A + m * K + k0 + 16 + half * 8);
  B32x2 r; r.a = *p0; r.b = *p1;
  return __builtin_bit_cast(v16bf, r);
}
__device__ __forceinline__ v16bf load_frag_b(const unsigned short* __restrict__ W,
                                             size_t n, int K, int k0, int half) {
  // B column n = W row n; lanes<16 hold K=[0..7],[16..23]; lanes>=16 hold K=[8..15],[24..31]
  const uint4* p0 = reinterpret_cast<const uint4*>(W + n * K + k0 + half * 8);
  const uint4* p1 = reinterpret_cast<const uint4*>(W + n * K + k0 + 16 + half * 8);
  B32x2 r; r.a = *p0; r.b = *p1;
  return __builtin_bit_cast(v16bf, r);
}

// ---------------- generic bf16 WMMA GEMM: C(MxN) = act(A(MxK) * W(NxK)^T + bias) ----------------
// 32x32 macro-tile per wave (MT=2 x NT=2): 32 acc + 16 A + 16 B VGPRs -> no spills, compiler can
// double-buffer fragments across k-steps (staged s_wait_loadcnt) without scratch traffic.
#define GEMM_NT 2
#define GEMM_MT 2
__global__ void k_gemm_bf16(const unsigned short* __restrict__ A,
                            const unsigned short* __restrict__ W,
                            const float* __restrict__ bias,
                            float* __restrict__ Cf,
                            unsigned short* __restrict__ Cb,
                            int M, int N, int K, int act)
{
  int wave = (blockIdx.x * blockDim.x + threadIdx.x) >> 5;
  int lane = threadIdx.x & 31;
  int half = lane >> 4;          // 0: lanes 0-15, 1: lanes 16-31
  int col  = lane & 15;
  int tn = N / (16 * GEMM_NT);
  int m0 = (wave / tn) * (16 * GEMM_MT);
  int n0 = (wave % tn) * (16 * GEMM_NT);
  if (m0 >= M) return;           // wave-uniform; grids are exact so never taken

  size_t mrow[GEMM_MT];
#pragma unroll
  for (int i = 0; i < GEMM_MT; ++i) mrow[i] = (size_t)(m0 + 16 * i + col);
  size_t ncol[GEMM_NT];
#pragma unroll
  for (int j = 0; j < GEMM_NT; ++j) ncol[j] = (size_t)(n0 + 16 * j + col);

  v8f acc[GEMM_MT][GEMM_NT] = {};
  for (int k0 = 0; k0 < K; k0 += 32) {
    // ---- batch all fragment loads (one clause, staged waits) ----
    v16bf af[GEMM_MT];
#pragma unroll
    for (int i = 0; i < GEMM_MT; ++i) af[i] = load_frag_a(A, mrow[i], K, k0, half);
    v16bf bf[GEMM_NT];
#pragma unroll
    for (int j = 0; j < GEMM_NT; ++j) bf[j] = load_frag_b(W, ncol[j], K, k0, half);
    // ---- batch WMMAs ----
#pragma unroll
    for (int j = 0; j < GEMM_NT; ++j) {
#pragma unroll
      for (int i = 0; i < GEMM_MT; ++i)
        acc[i][j] = __builtin_amdgcn_wmma_f32_16x16x32_bf16(false, af[i], false, bf[j],
                                                            (short)0, acc[i][j], false, false);
    }
  }
  // Epilogue: C/D layout — VGPR v holds row (tile_m0 + v + half*8), column (tile_n0 + col)
#pragma unroll
  for (int i = 0; i < GEMM_MT; ++i) {
    int m_base = m0 + 16 * i + half * 8;
#pragma unroll
    for (int j = 0; j < GEMM_NT; ++j) {
      int n = n0 + 16 * j + col;
      float bv = bias ? bias[n] : 0.0f;
#pragma unroll
      for (int v = 0; v < 8; ++v) {
        float val = acc[i][j][v] + bv;
        if (act == 1) val = tanhf(val);
        size_t idx = (size_t)(m_base + v) * N + n;
        if (Cf) Cf[idx] = val;
        if (Cb) Cb[idx] = f32_to_bf16(val);
      }
    }
  }
}

// ---------------- prep kernels ----------------
__global__ void k_f32_to_bf16(const float* __restrict__ s, unsigned short* __restrict__ d, int n) {
  int i = blockIdx.x * blockDim.x + threadIdx.x;
  if (i < n) d[i] = f32_to_bf16(s[i]);
}
// Wc (2048 x 1536) = [W_ih | W_hh] in bf16
__global__ void k_build_wc(const float* __restrict__ wih, const float* __restrict__ whh,
                           unsigned short* __restrict__ wc) {
  int i = blockIdx.x * blockDim.x + threadIdx.x;
  if (i >= 4 * HH * (EE + 2 * HH)) return;
  int n = i / (EE + 2 * HH), k = i % (EE + 2 * HH);
  float v = (k < EE + HH) ? wih[(size_t)n * (EE + HH) + k]
                          : whh[(size_t)n * HH + (k - EE - HH)];
  wc[i] = f32_to_bf16(v);
}
__global__ void k_bias(const float* __restrict__ a, const float* __restrict__ b,
                       float* __restrict__ o) {
  int i = blockIdx.x * blockDim.x + threadIdx.x;
  if (i < 4 * HH) o[i] = a[i] + b[i];
}
__global__ void k_init_state(const float* __restrict__ h0, const float* __restrict__ c0,
                             float* __restrict__ h, float* __restrict__ c,
                             unsigned short* __restrict__ hb) {
  int i = blockIdx.x * blockDim.x + threadIdx.x;
  if (i < BB * HH) { h[i] = h0[i]; c[i] = c0[i]; hb[i] = f32_to_bf16(h0[i]); }
}

// ---------------- per-step kernels ----------------
// Xc (B x 1536) = [embed(tgt_t) | attn_h(t-1) | h(t-1)] in bf16
__global__ void k_build_xc(const float* __restrict__ emb, const int* __restrict__ tgt,
                           const unsigned short* __restrict__ hist,
                           const unsigned short* __restrict__ hb,
                           unsigned short* __restrict__ Xc, int t)
{
  int i = blockIdx.x * blockDim.x + threadIdx.x;
  if (i >= BB * (EE + 2 * HH)) return;
  int b = i / (EE + 2 * HH), k = i % (EE + 2 * HH);
  unsigned short v;
  if (k < EE) {
    int tok = tgt[t * BB + b];
    v = f32_to_bf16(emb[(size_t)tok * EE + k]);
  } else if (k < EE + HH) {
    v = t ? hist[((size_t)(t - 1) * BB + b) * HH + (k - EE)] : (unsigned short)0;
  } else {
    v = hb[b * HH + (k - EE - HH)];
  }
  Xc[i] = v;
}

// LSTM cell (PyTorch gate order i,f,g,o)
__global__ void k_cell(const float* __restrict__ gates, float* __restrict__ c,
                       float* __restrict__ h, unsigned short* __restrict__ hb)
{
  int i = blockIdx.x * blockDim.x + threadIdx.x;
  if (i >= BB * HH) return;
  int b = i >> 9, j = i & (HH - 1);
  const float* g = gates + (size_t)b * 4 * HH;
  float ig = g[j], fg = g[j + HH], gg = g[j + 2 * HH], og = g[j + 3 * HH];
  float cn = sigm(fg) * c[i] + sigm(ig) * tanhf(gg);
  float hn = sigm(og) * tanhf(cn);
  c[i] = cn; h[i] = hn; hb[i] = f32_to_bf16(hn);
}

// Attention: scores over S, softmax, context; builds Xa = [c_vec | h_t] bf16. One block per batch b.
__global__ void k_attn(const float* __restrict__ q, const float* __restrict__ src,
                       const unsigned short* __restrict__ hb,
                       unsigned short* __restrict__ Xa,
                       float* __restrict__ out_align, int t)
{
  __shared__ float sc[SS];
  __shared__ float red[256];
  int b = blockIdx.x, tid = threadIdx.x;
  const float* qrow = q + (size_t)b * HH;
  for (int s = tid; s < SS; s += 256) {
    const float* crow = src + ((size_t)s * BB + b) * HH;
    float d = 0.0f;
    for (int k = 0; k < HH; k += 4) {
      float4 cv = *reinterpret_cast<const float4*>(crow + k);
      float4 qv = *reinterpret_cast<const float4*>(qrow + k);
      d += cv.x * qv.x + cv.y * qv.y + cv.z * qv.z + cv.w * qv.w;
    }
    sc[s] = d;
  }
  __syncthreads();
  red[tid] = fmaxf(sc[tid], sc[tid + 256]);
  __syncthreads();
  for (int off = 128; off > 0; off >>= 1) {
    if (tid < off) red[tid] = fmaxf(red[tid], red[tid + off]);
    __syncthreads();
  }
  float mx = red[0];
  __syncthreads();
  float e0 = __expf(sc[tid] - mx), e1 = __expf(sc[tid + 256] - mx);
  red[tid] = e0 + e1;
  __syncthreads();
  for (int off = 128; off > 0; off >>= 1) {
    if (tid < off) red[tid] += red[tid + off];
    __syncthreads();
  }
  float inv = 1.0f / red[0];
  __syncthreads();
  sc[tid] = e0 * inv; sc[tid + 256] = e1 * inv;
  __syncthreads();
  for (int k = tid; k < HH; k += 256) {
    float acc = 0.0f;
    for (int s = 0; s < SS; ++s) acc += sc[s] * src[((size_t)s * BB + b) * HH + k];
    Xa[b * 2 * HH + k] = f32_to_bf16(acc);
    Xa[b * 2 * HH + HH + k] = hb[b * HH + k];
  }
  if (t == TT - 1)
    for (int s = tid; s < SS; s += 256) out_align[b * SS + s] = sc[s];
}

__global__ void k_final(const float* __restrict__ h, const float* __restrict__ c,
                        float* __restrict__ out_h, float* __restrict__ out_c)
{
  int i = blockIdx.x * blockDim.x + threadIdx.x;
  if (i < BB * HH) { out_h[i] = h[i]; out_c[i] = c[i]; }
}

// ---------------- host launcher ----------------
extern "C" void kernel_launch(void* const* d_in, const int* in_sizes, int n_in,
                              void* d_out, int out_size, void* d_ws, size_t ws_size,
                              hipStream_t stream) {
  const int*   tgt   = (const int*)  d_in[0];
  const float* srcE  = (const float*)d_in[1];
  const float* h0    = (const float*)d_in[2];
  const float* c0    = (const float*)d_in[3];
  const float* emb   = (const float*)d_in[4];
  const float* Wih   = (const float*)d_in[5];
  const float* bih   = (const float*)d_in[6];
  const float* Whh   = (const float*)d_in[7];
  const float* bhh   = (const float*)d_in[8];
  const float* Wat   = (const float*)d_in[9];
  const float* Wout  = (const float*)d_in[10];
  const float* Wproj = (const float*)d_in[11];
  float* out = (float*)d_out;

  const size_t OFF_H  = (size_t)TT * BB * VV;
  const size_t OFF_C  = OFF_H + BB * HH;
  const size_t OFF_AL = OFF_C + BB * HH;

  char* p = (char*)d_ws;
  auto carve = [&](size_t bytes) -> char* {
    char* r = p; p += (bytes + 255) & ~(size_t)255; return r;
  };
  const int KC = EE + 2 * HH;  // 1536
  unsigned short* wc_b    = (unsigned short*)carve((size_t)4 * HH * KC * 2);
  unsigned short* wat_b   = (unsigned short*)carve((size_t)HH * HH * 2);
  unsigned short* wout_b  = (unsigned short*)carve((size_t)HH * 2 * HH * 2);
  unsigned short* wproj_b = (unsigned short*)carve((size_t)VV * HH * 2);
  float*          bias    = (float*)         carve((size_t)4 * HH * 4);
  unsigned short* Xc      = (unsigned short*)carve((size_t)BB * KC * 2);
  float*          gates   = (float*)         carve((size_t)BB * 4 * HH * 4);
  float*          hbuf    = (float*)         carve((size_t)BB * HH * 4);
  float*          cbuf    = (float*)         carve((size_t)BB * HH * 4);
  unsigned short* hb16    = (unsigned short*)carve((size_t)BB * HH * 2);
  float*          qbuf    = (float*)         carve((size_t)BB * HH * 4);
  unsigned short* Xa      = (unsigned short*)carve((size_t)BB * 2 * HH * 2);
  unsigned short* hist    = (unsigned short*)carve((size_t)TT * BB * HH * 2);

  auto gemm = [&](const unsigned short* A, const unsigned short* W, const float* bi,
                  float* Cf, unsigned short* Cb, int M, int N, int K, int act) {
    int waves = (M / 32) * (N / 32);           // 32x32 macro-tile per wave
    int blocks = (waves + 7) / 8;              // 8 waves (256 threads) per WG
    k_gemm_bf16<<<blocks, 256, 0, stream>>>(A, W, bi, Cf, Cb, M, N, K, act);
  };
  auto ew = [&](int n) { return dim3((n + 255) / 256); };

  // one-time prep (re-run every call: deterministic)
  k_build_wc<<<ew(4 * HH * KC), 256, 0, stream>>>(Wih, Whh, wc_b);
  k_f32_to_bf16<<<ew(HH * HH), 256, 0, stream>>>(Wat, wat_b, HH * HH);
  k_f32_to_bf16<<<ew(HH * 2 * HH), 256, 0, stream>>>(Wout, wout_b, HH * 2 * HH);
  k_f32_to_bf16<<<ew(VV * HH), 256, 0, stream>>>(Wproj, wproj_b, VV * HH);
  k_bias<<<ew(4 * HH), 256, 0, stream>>>(bih, bhh, bias);
  k_init_state<<<ew(BB * HH), 256, 0, stream>>>(h0, c0, hbuf, cbuf, hb16);

  for (int t = 0; t < TT; ++t) {
    k_build_xc<<<ew(BB * KC), 256, 0, stream>>>(emb, tgt, hist, hb16, Xc, t);
    gemm(Xc, wc_b, bias, gates, nullptr, BB, 4 * HH, KC, 0);          // gates
    k_cell<<<ew(BB * HH), 256, 0, stream>>>(gates, cbuf, hbuf, hb16); // LSTM cell
    gemm(hb16, wat_b, nullptr, qbuf, nullptr, BB, HH, HH, 0);         // q = h W_attn^T
    k_attn<<<BB, 256, 0, stream>>>(qbuf, srcE, hb16, Xa, out + OFF_AL, t);
    gemm(Xa, wout_b, nullptr, nullptr, hist + (size_t)t * BB * HH,    // attn_h (tanh, bf16)
         BB, HH, 2 * HH, 1);
  }

  // big vocab projection: (T*B x H) x (H x V) -> scores, straight into d_out
  gemm(hist, wproj_b, nullptr, out, nullptr, TT * BB, VV, HH, 0);
  k_final<<<ew(BB * HH), 256, 0, stream>>>(hbuf, cbuf, out + OFF_H, out + OFF_C);
}
